// MultiHeadedAttention_5944234738147
// MI455X (gfx1250) — matrix operand order repaired and checked by reading, not compiled, not run on hardware
//
#include <hip/hip_runtime.h>

// ---------------- problem constants ----------------
constexpr int HEAD_SIZE = 64;
constexpr int NUM_HEADS = 12;
constexpr int SEQ_LEN   = 2048;
constexpr int BATCH     = 4;
constexpr int D_MODEL   = NUM_HEADS * HEAD_SIZE;   // 768
constexpr int M_TOT     = BATCH * SEQ_LEN;         // 8192

using half_t = _Float16;
typedef __attribute__((ext_vector_type(16))) _Float16 v16h;
typedef __attribute__((ext_vector_type(8)))  _Float16 v8h;
typedef __attribute__((ext_vector_type(8)))  float    v8f;

#define DEV static __device__ __forceinline__

DEV v8f zero8() {
    v8f z;
#pragma unroll
    for (int i = 0; i < 8; ++i) z[i] = 0.0f;
    return z;
}

DEV v8f wmma_f16(v16h a, v16h b, v8f c) {
    // D = A(16x32 f16) * B(32x16 f16) + C(16x16 f32)
    return __builtin_amdgcn_wmma_f32_16x16x32_f16(
        false, a, false, b, (short)0, c, false, false);
}

// Low 32 bits of a flat LDS pointer == LDS byte offset (ISA 10.2).
DEV unsigned to_lds(const void* p) { return (unsigned)(unsigned long long)p; }

// Async copy 16B per lane: global -> LDS. Tracked by ASYNCcnt.
DEV void async_b128(const void* gaddr, unsigned lds_byte_addr) {
    asm volatile("global_load_async_to_lds_b128 %0, %1, off"
                 :: "v"(lds_byte_addr), "v"(gaddr) : "memory");
}
DEV void wait_async_le2() { asm volatile("s_wait_asynccnt 2" ::: "memory"); }
DEV void wait_async_le4() { asm volatile("s_wait_asynccnt 4" ::: "memory"); }
DEV void wait_async_0()   { asm volatile("s_wait_asynccnt 0" ::: "memory"); }

// A fragment 16x32 f16. `base` -> element [row0][k0]; row stride ldk halves.
// lanes 0-15 row=lane: e0..7=K0..7, e8..15=K16..23;
// lanes 16-31 row=lane-16: e0..7=K8..15, e8..15=K24..31.
DEV v16h load_a32(const half_t* base, int ldk) {
    const int lane = threadIdx.x & 31;
    const int r    = lane & 15;
    const int ko   = (lane & 16) ? 8 : 0;
    const half_t* p = base + (size_t)r * ldk + ko;
    v8h c0 = *(const v8h*)(p);
    v8h c1 = *(const v8h*)(p + 16);
    return __builtin_shufflevector(c0, c1, 0,1,2,3,4,5,6,7,8,9,10,11,12,13,14,15);
}

// B fragment 32x16 f16; B's column n contiguous at base + n*ldn + k.
// lanes 0-15 col=lane: e0..7=K0..7, e8..15=K16..23; lanes 16-31: e0..7=K8..15, e8..15=K24..31.
DEV v16h load_b32(const half_t* base, int ldn) {
    const int lane = threadIdx.x & 31;
    const int n    = lane & 15;
    const int ko   = (lane & 16) ? 8 : 0;
    const half_t* p = base + (size_t)n * ldn + ko;
    v8h c0 = *(const v8h*)(p);
    v8h c1 = *(const v8h*)(p + 16);
    return __builtin_shufflevector(c0, c1, 0,1,2,3,4,5,6,7,8,9,10,11,12,13,14,15);
}

// ---------------- f32 -> f16 conversion ----------------
__global__ __launch_bounds__(256) void cvt_f32_f16(const float* __restrict__ in,
                                                   half_t* __restrict__ out,
                                                   size_t n) {
    size_t i = (size_t)blockIdx.x * blockDim.x + threadIdx.x;
    size_t stride = (size_t)gridDim.x * blockDim.x;
    for (; i < n; i += stride) out[i] = (half_t)in[i];
}

// ---------------- WMMA GEMM:  out = A(MxK) * W(NxK)^T + bias ----------------
// Block: 4 waves, 128 (m) x 64 (n) output; wave: 32x64.
// W tile (64 n x 32 k) double-buffered in LDS via async global->LDS.
// mode 0: f32 out [M][N] ; 1: Q f16 [B][H][S][64] *0.125 ; 2: K f16 same ; 3: V^T f16 [B][H][64][S]
__global__ __launch_bounds__(128) void gemm_wmma(const half_t* __restrict__ A,
                                                 const half_t* __restrict__ W,
                                                 const float* __restrict__ bias,
                                                 void* __restrict__ out,
                                                 int M, int N, int K, int mode) {
    __shared__ __align__(16) half_t wtile[2][64 * 32];  // 4KB per buffer

    const int tid  = threadIdx.x;
    const int wave = tid >> 5;
    const int lane = tid & 31;
    const int m0   = blockIdx.x * 128 + wave * 32;
    const int n0   = blockIdx.y * 64;

    auto prefetch = [&](int kk, int buf) {
#pragma unroll
        for (int u = 0; u < 2; ++u) {
            const int c    = tid + u * 128;          // 256 x 16B chunks
            const int row  = c >> 2;                 // 64 rows
            const int colh = (c & 3) * 8;            // 4 chunks of 8 halves
            async_b128(W + (size_t)(n0 + row) * K + kk + colh,
                       to_lds(&wtile[buf][row * 32 + colh]));
        }
    };

    v8f acc[2][4];
#pragma unroll
    for (int mi = 0; mi < 2; ++mi)
#pragma unroll
        for (int j = 0; j < 4; ++j) acc[mi][j] = zero8();

    const int steps = K / 32;
    prefetch(0, 0);
    for (int t = 0; t < steps; ++t) {
        const int buf = t & 1;
        if (t + 1 < steps) { prefetch((t + 1) * 32, buf ^ 1); wait_async_le2(); }
        else               { wait_async_0(); }
        __syncthreads();   // everyone's waits done -> tile visible

        v16h a0 = load_a32(A + (size_t)(m0)      * K + t * 32, K);
        v16h a1 = load_a32(A + (size_t)(m0 + 16) * K + t * 32, K);
#pragma unroll
        for (int j = 0; j < 4; ++j) {
            v16h b = load_b32(&wtile[buf][(j * 16) * 32], 32);
            acc[0][j] = wmma_f16(a0, b, acc[0][j]);
            acc[1][j] = wmma_f16(a1, b, acc[1][j]);
        }
        __syncthreads();   // reads done before buffer is async-overwritten
    }

    const int col = lane & 15;
    const int rb  = (lane & 16) ? 8 : 0;
    const float qscale = 0.125f; // 1/sqrt(64)

#pragma unroll
    for (int mi = 0; mi < 2; ++mi) {
#pragma unroll
        for (int j = 0; j < 4; ++j) {
            const int n = n0 + j * 16 + col;
            const float bv = bias[n];
#pragma unroll
            for (int i = 0; i < 8; ++i) {
                const int m = m0 + mi * 16 + rb + i;
                float val = acc[mi][j][i] + bv;
                if (mode == 0) {
                    ((float*)out)[(size_t)m * N + n] = val;
                } else {
                    const int b = m / SEQ_LEN, s = m % SEQ_LEN;
                    const int h = n / HEAD_SIZE, d = n % HEAD_SIZE;
                    half_t* o = (half_t*)out;
                    if (mode == 1) {
                        o[(((size_t)(b * NUM_HEADS + h) * SEQ_LEN) + s) * HEAD_SIZE + d] =
                            (half_t)(val * qscale);
                    } else if (mode == 2) {
                        o[(((size_t)(b * NUM_HEADS + h) * SEQ_LEN) + s) * HEAD_SIZE + d] =
                            (half_t)val;
                    } else {
                        o[(((size_t)(b * NUM_HEADS + h) * HEAD_SIZE) + d) * SEQ_LEN + s] =
                            (half_t)val;
                    }
                }
            }
        }
    }
}

// ---------------- flash attention core ----------------
// All 4 waves of a block share (b,h): K tile (32x64) and V tile (64x32) are
// double-buffered in LDS via async global->LDS; P tile is per-wave LDS.
__global__ __launch_bounds__(128) void attn_wmma(const half_t* __restrict__ Q,
                                                 const half_t* __restrict__ Kr,
                                                 const half_t* __restrict__ Vt,
                                                 const long long* __restrict__ mask,
                                                 half_t* __restrict__ Aout) {
    __shared__ __align__(16) half_t ktile[2][32 * 64];  // keys x head-dim, 4KB
    __shared__ __align__(16) half_t vtile[2][64 * 32];  // head-dim x keys, 4KB
    __shared__ __align__(16) half_t ptile[4][16 * 32];  // per-wave P

    const int tid  = threadIdx.x;
    const int wave = tid >> 5;
    const int lane = tid & 31;
    const int bh   = blockIdx.x >> 5;                 // 32 blocks per (b,h)
    const int qt   = (blockIdx.x & 31) * 4 + wave;
    const int b    = bh / NUM_HEADS;
    const int h    = bh % NUM_HEADS;
    const int q0   = qt * 16;
    const int col  = lane & 15;
    const int rb   = (lane & 16) ? 8 : 0;

    const half_t* Kbase = Kr + (size_t)bh * SEQ_LEN * HEAD_SIZE;
    const half_t* Vbase = Vt + (size_t)bh * HEAD_SIZE * SEQ_LEN;

    auto prefetch = [&](int kv, int buf) {
#pragma unroll
        for (int u = 0; u < 2; ++u) {
            const int c = tid + u * 128;              // 256 chunks each tile
            // K tile: 32 rows x 128B (8 chunks/row)
            const int kr = c >> 3, kc = (c & 7) * 8;
            async_b128(Kbase + (size_t)(kv + kr) * HEAD_SIZE + kc,
                       to_lds(&ktile[buf][kr * 64 + kc]));
            // V tile: 64 rows x 64B (4 chunks/row)
            const int vr = c >> 2, vc = (c & 3) * 8;
            async_b128(Vbase + (size_t)vr * SEQ_LEN + kv + vc,
                       to_lds(&vtile[buf][vr * 32 + vc]));
        }
    };

    const half_t* Qb = Q + ((size_t)bh * SEQ_LEN + q0) * HEAD_SIZE;
    const v16h qa0 = load_a32(Qb,      HEAD_SIZE);
    const v16h qa1 = load_a32(Qb + 32, HEAD_SIZE);

    const float NEG = -1.0e30f;
    float mrow[8], lrow[8];
    v8f o[4];
#pragma unroll
    for (int i = 0; i < 8; ++i) { mrow[i] = NEG; lrow[i] = 0.0f; }
#pragma unroll
    for (int j = 0; j < 4; ++j) o[j] = zero8();

    half_t* P = &ptile[wave][0];

    prefetch(0, 0);
    for (int kv = 0; kv < SEQ_LEN; kv += 32) {
        const int buf = (kv >> 5) & 1;
        if (kv + 32 < SEQ_LEN) { prefetch(kv + 32, buf ^ 1); wait_async_le4(); }
        else                   { wait_async_0(); }
        __syncthreads();

        // ---- scores: two 16x16 tiles from LDS K tile ----
        v8f s0 = wmma_f16(qa0, load_b32(&ktile[buf][0],           64), zero8());
        s0     = wmma_f16(qa1, load_b32(&ktile[buf][32],          64), s0);
        v8f s1 = wmma_f16(qa0, load_b32(&ktile[buf][16 * 64],     64), zero8());
        s1     = wmma_f16(qa1, load_b32(&ktile[buf][16 * 64 + 32],64), s1);

        // ---- mask (per-lane key column) ----
        const long long mk0 = mask[(size_t)b * SEQ_LEN + kv + col];
        const long long mk1 = mask[(size_t)b * SEQ_LEN + kv + 16 + col];
        if (mk0 == 0) {
#pragma unroll
            for (int i = 0; i < 8; ++i) s0[i] = NEG;
        }
        if (mk1 == 0) {
#pragma unroll
            for (int i = 0; i < 8; ++i) s1[i] = NEG;
        }

        // ---- online softmax (row i across a 16-lane group) ----
#pragma unroll
        for (int i = 0; i < 8; ++i) {
            float t = fmaxf(s0[i], s1[i]);
#pragma unroll
            for (int off = 1; off < 16; off <<= 1)
                t = fmaxf(t, __shfl_xor(t, off, 32));
            const float mn    = fmaxf(mrow[i], t);
            const float alpha = __expf(mrow[i] - mn);
            mrow[i] = mn;
            const float p0 = __expf(s0[i] - mn);
            const float p1 = __expf(s1[i] - mn);
            s0[i] = p0; s1[i] = p1;
            float rs = p0 + p1;
#pragma unroll
            for (int off = 1; off < 16; off <<= 1)
                rs += __shfl_xor(rs, off, 32);
            lrow[i] = lrow[i] * alpha + rs;
#pragma unroll
            for (int j = 0; j < 4; ++j) o[j][i] *= alpha;
        }

        // ---- C-layout P -> per-wave LDS -> A-layout fragment ----
#pragma unroll
        for (int i = 0; i < 8; ++i) {
            P[(rb + i) * 32 + col]      = (half_t)s0[i];
            P[(rb + i) * 32 + 16 + col] = (half_t)s1[i];
        }
        asm volatile("" ::: "memory");  // intra-wave LDS exchange (lockstep wave32)
        const v16h pa = load_a32(P, 32);

        // ---- O += P(16x32) * V(32x64) from LDS V tile ----
#pragma unroll
        for (int j = 0; j < 4; ++j) {
            v16h vb = load_b32(&vtile[buf][(j * 16) * 32], 32);
            o[j] = wmma_f16(pa, vb, o[j]);
        }
        __syncthreads();  // tile reads done before async overwrite
    }

    // ---- normalize, store merged-heads [B][S][D_MODEL] ----
#pragma unroll
    for (int j = 0; j < 4; ++j) {
#pragma unroll
        for (int i = 0; i < 8; ++i) {
            const int s = q0 + rb + i;
            const int d = h * HEAD_SIZE + j * 16 + col;
            Aout[((size_t)b * SEQ_LEN + s) * D_MODEL + d] =
                (half_t)(o[j][i] / lrow[i]);
        }
    }
}

// ---------------- launcher ----------------
extern "C" void kernel_launch(void* const* d_in, const int* in_sizes, int n_in,
                              void* d_out, int out_size, void* d_ws, size_t ws_size,
                              hipStream_t stream) {
    const float*     X    = (const float*)d_in[0];
    const long long* mask = (const long long*)d_in[1];
    const float*     Wq   = (const float*)d_in[2];
    const float*     bq   = (const float*)d_in[3];
    const float*     Wk   = (const float*)d_in[4];
    const float*     bk   = (const float*)d_in[5];
    const float*     Wv   = (const float*)d_in[6];
    const float*     bv   = (const float*)d_in[7];
    const float*     Wy   = (const float*)d_in[8];
    const float*     by   = (const float*)d_in[9];

    constexpr size_t SX = (size_t)M_TOT * D_MODEL;      // 8192*768
    constexpr size_t SW = (size_t)D_MODEL * D_MODEL;    // 768*768
    char* ws = (char*)d_ws;
    half_t* Xh  = (half_t*)(ws);
    half_t* Wqh = (half_t*)(ws + SX * 2);
    half_t* Wkh = Wqh + SW;
    half_t* Wvh = Wkh + SW;
    half_t* Wyh = Wvh + SW;
    half_t* Qh  = Wyh + SW;                 // [B][H][S][64]
    half_t* Kh  = Qh + SX;                  // [B][H][S][64]
    half_t* Vt  = Kh + SX;                  // [B][H][64][S]
    half_t* Ah  = Vt + SX;                  // merged [B][S][768]

    cvt_f32_f16<<<2048, 256, 0, stream>>>(X,  Xh,  SX);
    cvt_f32_f16<<<512,  256, 0, stream>>>(Wq, Wqh, SW);
    cvt_f32_f16<<<512,  256, 0, stream>>>(Wk, Wkh, SW);
    cvt_f32_f16<<<512,  256, 0, stream>>>(Wv, Wvh, SW);
    cvt_f32_f16<<<512,  256, 0, stream>>>(Wy, Wyh, SW);

    dim3 ggrid(M_TOT / 128, D_MODEL / 64);
    gemm_wmma<<<ggrid, 128, 0, stream>>>(Xh, Wqh, bq, Qh, M_TOT, D_MODEL, D_MODEL, 1);
    gemm_wmma<<<ggrid, 128, 0, stream>>>(Xh, Wkh, bk, Kh, M_TOT, D_MODEL, D_MODEL, 2);
    gemm_wmma<<<ggrid, 128, 0, stream>>>(Xh, Wvh, bv, Vt, M_TOT, D_MODEL, D_MODEL, 3);

    attn_wmma<<<(BATCH * NUM_HEADS * (SEQ_LEN / 16)) / 4, 128, 0, stream>>>(
        Qh, Kh, Vt, mask, Ah);

    gemm_wmma<<<ggrid, 128, 0, stream>>>(Ah, Wyh, by, d_out, M_TOT, D_MODEL, D_MODEL, 0);
}
